// ContinuousThoughtBlock_41970420416583
// MI455X (gfx1250) — matrix operand order repaired and checked by reading, not compiled, not run on hardware
//
#include <hip/hip_runtime.h>
#include <math.h>

// ---------------------------------------------------------------------------
// ContinuousThoughtBlock for MI455X (gfx1250, wave32, WMMA bf16 + TDM async)
// ---------------------------------------------------------------------------

typedef __attribute__((ext_vector_type(16))) __bf16 bf16x16;
typedef __attribute__((ext_vector_type(8)))  __bf16 bf16x8;
typedef __attribute__((ext_vector_type(8)))  float  f32x8;
typedef __attribute__((ext_vector_type(4)))  unsigned int u32x4;
typedef __attribute__((ext_vector_type(8)))  int i32x8;
typedef __attribute__((ext_vector_type(4)))  int i32x4;

#define B_DIM 8
#define L_DIM 2048
#define D_DIM 1024
#define H_DIM 4096
#define NROWS 64   // NUM_PATHS * B

// Tensor Data Mover availability (this toolchain: 6-arg builtin,
// (u32x4 g0, i32x8 g1, i32x4 g2, i32x4 g3, i32x8 g4, i32 cpol))
#if defined(__has_builtin)
#if __has_builtin(__builtin_amdgcn_tensor_load_to_lds) && \
    __has_builtin(__builtin_amdgcn_s_wait_tensorcnt)
#define HAVE_TDM 1
#endif
#endif
#ifndef HAVE_TDM
#define HAVE_TDM 0
#endif

// ---------------- helpers ----------------

__device__ __forceinline__ unsigned short f2bf(float f) {
  unsigned int u = __float_as_uint(f);
  u += 0x7FFFu + ((u >> 16) & 1u);   // round-to-nearest-even
  return (unsigned short)(u >> 16);
}

__device__ __forceinline__ f32x8 zero8() {
  f32x8 z;
#pragma unroll
  for (int i = 0; i < 8; ++i) z[i] = 0.0f;
  return z;
}

// A fragment (16x32 bf16, row-major source, stride in elements).
// ISA layout: lanes 0-15 hold row M=lane, K {kb..kb+7, kb+16..kb+23};
// lanes 16-31 hold row M=lane-16, K {kb+8..kb+15, kb+24..kb+31}.
__device__ __forceinline__ bf16x16 load_a_frag(const unsigned short* base, int stride,
                                               int mrow, int kbase, int lane) {
  const int r  = mrow + (lane & 15);
  const int k0 = kbase + ((lane & 16) ? 8 : 0);
  union { bf16x16 v; bf16x8 h[2]; } u;
  u.h[0] = *(const bf16x8*)(base + (size_t)r * stride + k0);
  u.h[1] = *(const bf16x8*)(base + (size_t)r * stride + k0 + 16);
  return u.v;
}

// B fragment (32x16 bf16) from K-contiguous (transposed) weights Wt[n][k].
// ISA layout: lanes 0-15: N=lane, K {kb..kb+7, kb+16..kb+23}; lanes 16-31: N=lane-16, K {kb+8..kb+15, kb+24..kb+31}.
__device__ __forceinline__ bf16x16 load_b_frag(const unsigned short* baseT, int stride,
                                               int nbase, int kbase, int lane) {
  const int n  = nbase + (lane & 15);
  const int k0 = kbase + ((lane & 16) ? 8 : 0);
  union { bf16x16 v; bf16x8 h[2]; } u;
  u.h[0] = *(const bf16x8*)(baseT + (size_t)n * stride + k0);
  u.h[1] = *(const bf16x8*)(baseT + (size_t)n * stride + k0 + 16);
  return u.v;
}

__device__ __forceinline__ f32x8 wmma_bf16f32(bf16x16 a, bf16x16 b, f32x8 c) {
  return __builtin_amdgcn_wmma_f32_16x16x32_bf16(false, a, false, b, (short)0, c,
                                                 false, false);
}

#if HAVE_TDM
// Issue a TDM 2D tile load: 256 rows x 64 bytes from bf16 Wt[n][k]
// (global row stride 2048 B) into LDS at lds_off with 16 B padding after
// every 64 B (=> padded LDS row stride 80 B == 40 bf16).
// D# encoding per CDNA5 ISA ch.8 (group0: 4 dw, group1: 8 dw).
__device__ __forceinline__ void tdm_issue_b(const unsigned short* gsrc,
                                            unsigned lds_off) {
  const unsigned long long ga = (unsigned long long)(size_t)gsrc;
  u32x4 g0;
  g0[0] = 1u;                                            // count=1, user mode
  g0[1] = lds_off;                                       // lds_addr (bytes)
  g0[2] = (unsigned)(ga & 0xFFFFFFFFu);                  // global_addr[31:0]
  g0[3] = (unsigned)((ga >> 32) & 0x01FFFFFFu)           // global_addr[56:32]
          | 0x80000000u;                                 // type=2 ("image")
  i32x8 g1;
  g1[0] = (3 << 16)      // data_size = 3 (8 bytes)
        | (1 << 20)      // pad_enable
        | (3 << 22)      // pad_interval: 16 DWORDs = 64 B
        | (3 << 25);     // pad_amount:   4 DWORDs = 16 B
  g1[1] = (int)(256u << 16);      // tensor_dim0[15:0] = 256 (8B units)
  g1[2] = (int)0xFFFF0000u;       // tensor_dim0 hi=0 | tensor_dim1[15:0]=0xFFFF
  g1[3] = (8 << 16);              // tensor_dim1 hi=0 | tile_dim0 = 8 (64 B)
  g1[4] = 256;                    // tile_dim1 = 256 rows | tile_dim2 = 0
  g1[5] = 256;                    // tensor_dim0_stride = 256 (8B units = 2048 B)
  g1[6] = 0;
  g1[7] = 0;
  i32x4 z4;
  z4[0] = 0; z4[1] = 0; z4[2] = 0; z4[3] = 0;
  i32x8 z8;
#pragma unroll
  for (int i = 0; i < 8; ++i) z8[i] = 0;
  __builtin_amdgcn_tensor_load_to_lds(g0, g1, z4, z4, z8, 0);
}
#endif

// ---------------- weight transpose + fp32 -> bf16 convert ----------------
// dst[n][k] = bf16(src[k][n]);  src is K x N row-major.
__global__ __launch_bounds__(256) void k_convw(const float* __restrict__ src,
                                               unsigned short* __restrict__ dst,
                                               int K, int N) {
  __shared__ float tile[32][33];
  const int tx = threadIdx.x & 31;
  const int ty = threadIdx.x >> 5;
  const int n0 = blockIdx.x * 32;
  const int k0 = blockIdx.y * 32;
#pragma unroll
  for (int i = 0; i < 32; i += 8)
    tile[ty + i][tx] = src[(size_t)(k0 + ty + i) * N + n0 + tx];
  __syncthreads();
#pragma unroll
  for (int i = 0; i < 32; i += 8)
    dst[(size_t)(n0 + ty + i) * K + k0 + tx] = f2bf(tile[tx][ty + i]);
}

// ---------------- context = mean over L ----------------
__global__ __launch_bounds__(256) void k_ctx_partial(const float* __restrict__ h,
                                                     float* __restrict__ partial) {
  const int c = blockIdx.x;   // 0..31 chunk of L
  const int b = blockIdx.y;   // 0..7
  const int t = threadIdx.x;
  const float* base = h + ((size_t)b * L_DIM + (size_t)c * 64) * D_DIM + t * 4;
  float s0 = 0.f, s1 = 0.f, s2 = 0.f, s3 = 0.f;
  for (int r = 0; r < 64; ++r) {
    float4 x = *(const float4*)(base + (size_t)r * D_DIM);
    s0 += x.x; s1 += x.y; s2 += x.z; s3 += x.w;
  }
  float4 o; o.x = s0; o.y = s1; o.z = s2; o.w = s3;
  *(float4*)(partial + (size_t)(b * 32 + c) * D_DIM + t * 4) = o;
}

__global__ __launch_bounds__(256) void k_ctx_reduce(const float* __restrict__ partial,
                                                    float* __restrict__ context) {
  const int b = blockIdx.x;
  const int t = threadIdx.x;
  float s0 = 0.f, s1 = 0.f, s2 = 0.f, s3 = 0.f;
  for (int c = 0; c < 32; ++c) {
    float4 x = *(const float4*)(partial + (size_t)(b * 32 + c) * D_DIM + t * 4);
    s0 += x.x; s1 += x.y; s2 += x.z; s3 += x.w;
  }
  const float inv = 1.0f / (float)L_DIM;
  float4 o; o.x = s0 * inv; o.y = s1 * inv; o.z = s2 * inv; o.w = s3 * inv;
  *(float4*)(context + (size_t)b * D_DIM + t * 4) = o;
}

// ---------------- row-wise LayerNorm (D=1024, one block per row) ----------------
template <bool BF16OUT>
__global__ __launch_bounds__(256) void k_ln_rows(const float* __restrict__ in,
                                                 const float* __restrict__ gamma,
                                                 const float* __restrict__ beta,
                                                 void* __restrict__ outp) {
  __shared__ float red[256];
  __shared__ float stat[2];
  const int r = blockIdx.x;
  const int t = threadIdx.x;
  const float* row = in + (size_t)r * D_DIM;
  float4 x = *(const float4*)(row + t * 4);
  float s = x.x + x.y + x.z + x.w;
  float q = x.x * x.x + x.y * x.y + x.z * x.z + x.w * x.w;
  red[t] = s; __syncthreads();
  for (int o = 128; o > 0; o >>= 1) { if (t < o) red[t] += red[t + o]; __syncthreads(); }
  if (t == 0) stat[0] = red[0];
  __syncthreads();
  red[t] = q; __syncthreads();
  for (int o = 128; o > 0; o >>= 1) { if (t < o) red[t] += red[t + o]; __syncthreads(); }
  if (t == 0) stat[1] = red[0];
  __syncthreads();
  const float mu  = stat[0] * (1.0f / D_DIM);
  const float var = stat[1] * (1.0f / D_DIM) - mu * mu;
  const float rs  = rsqrtf(var + 1e-6f);
  float xv[4] = {x.x, x.y, x.z, x.w};
#pragma unroll
  for (int i = 0; i < 4; ++i) {
    const int c = t * 4 + i;
    const float o2 = (xv[i] - mu) * rs * gamma[c] + beta[c];
    if (BF16OUT) ((unsigned short*)outp)[(size_t)r * D_DIM + c] = f2bf(o2);
    else         ((float*)outp)[(size_t)r * D_DIM + c] = o2;
  }
}

// ---------------- aggregator: thought0 = LN(ctx) @ Wa + ba, seed 8 paths ----------------
__global__ __launch_bounds__(256) void k_agg_seed(const float* __restrict__ ctxn,
                                                  const float* __restrict__ Wa,
                                                  const float* __restrict__ ba,
                                                  float* __restrict__ thoughts) {
  const int idx = blockIdx.x * 256 + threadIdx.x;   // 0..8191
  const int b = idx >> 10;
  const int n = idx & (D_DIM - 1);
  const float* cr = ctxn + (size_t)b * D_DIM;
  float s = ba[n];
  for (int k = 0; k < D_DIM; ++k) s += cr[k] * Wa[(size_t)k * D_DIM + n];
#pragma unroll
  for (int p = 0; p < 8; ++p)
    thoughts[(size_t)(p * 8 + b) * D_DIM + n] = s * (1.0f + 0.02f * (float)p);
}

// ---------------- broadcast: bc = final @ Wbc + bbc ----------------
__global__ __launch_bounds__(256) void k_bcast(const float* __restrict__ fin,
                                               const float* __restrict__ W,
                                               const float* __restrict__ bias,
                                               float* __restrict__ bc) {
  const int idx = blockIdx.x * 256 + threadIdx.x;
  const int b = idx >> 10;
  const int n = idx & (D_DIM - 1);
  const float* fr = fin + (size_t)b * D_DIM;
  float s = bias[n];
  for (int k = 0; k < D_DIM; ++k) s += fr[k] * W[(size_t)k * D_DIM + n];
  bc[idx] = s;
}

// ---------------- projector dense1: y1 = gelu(xln @ W1 + b1) (WMMA) ----------------
__global__ __launch_bounds__(256) void k_fc1(const unsigned short* __restrict__ xln,
                                             const unsigned short* __restrict__ Wt1,
                                             const float* __restrict__ b1,
                                             unsigned short* __restrict__ y1) {
  const int t = threadIdx.x;
  const int lane = t & 31;
  const int w = t >> 5;
  const int m = (w & 3) * 16;
  const int nb = blockIdx.x * 32 + (w >> 2) * 16;
  f32x8 acc = zero8();
  for (int ks = 0; ks < D_DIM / 32; ++ks) {
    const int kb = ks * 32;
    bf16x16 a  = load_a_frag(xln, D_DIM, m, kb, lane);
    bf16x16 bb = load_b_frag(Wt1, D_DIM, nb, kb, lane);
    acc = wmma_bf16f32(a, bb, acc);
  }
  const int cl = lane & 15;
  const int col = nb + cl;
  const float bv = b1[col];
  const int rbase = m + ((lane & 16) ? 8 : 0);
#pragma unroll
  for (int v = 0; v < 8; ++v) {
    const float x = acc[v] + bv;
    const float c = 0.7978845608f * (x + 0.044715f * x * x * x);
    const float g = 0.5f * x * (1.0f + tanhf(c));   // jax.nn.gelu tanh approx
    y1[(size_t)(rbase + v) * H_DIM + col] = f2bf(g);
  }
}

// ---------------- projector dense2: thoughts += y1 @ W2 + b2 (WMMA) ----------------
__global__ __launch_bounds__(256) void k_fc2(const unsigned short* __restrict__ y1,
                                             const unsigned short* __restrict__ Wt2,
                                             const float* __restrict__ b2,
                                             float* __restrict__ th) {
  const int t = threadIdx.x;
  const int lane = t & 31;
  const int w = t >> 5;
  const int m = (w & 3) * 16;
  const int nb = blockIdx.x * 32 + (w >> 2) * 16;
  f32x8 acc = zero8();
  for (int ks = 0; ks < H_DIM / 32; ++ks) {
    const int kb = ks * 32;
    bf16x16 a  = load_a_frag(y1, H_DIM, m, kb, lane);
    bf16x16 bb = load_b_frag(Wt2, H_DIM, nb, kb, lane);
    acc = wmma_bf16f32(a, bb, acc);
  }
  const int cl = lane & 15;
  const int col = nb + cl;
  const float bv = b2[col];
  const int rbase = m + ((lane & 16) ? 8 : 0);
#pragma unroll
  for (int v = 0; v < 8; ++v) {
    float* p = th + (size_t)(rbase + v) * D_DIM + col;
    *p += acc[v] + bv;
  }
}

// ---------------- path scoring -> softmax -> prune -> amps ----------------
__global__ __launch_bounds__(256) void k_score(const float* __restrict__ th,
                                               const float* __restrict__ context,
                                               float* __restrict__ amps) {
  __shared__ float part[64][4];
  __shared__ float sc[64];
  const int t = threadIdx.x;
  const int r = t >> 2;
  const int q = t & 3;
  const float* tr = th + (size_t)r * D_DIM + q * 256;
  const float* cx = context + (size_t)(r & 7) * D_DIM + q * 256;
  float s = 0.f;
  for (int i = 0; i < 256; ++i) s += tr[i] * cx[i];
  part[r][q] = s;
  __syncthreads();
  if (t < 64)
    sc[t] = (part[t][0] + part[t][1] + part[t][2] + part[t][3]) * (1.0f / 32.0f);
  __syncthreads();
  if (t < 8) {
    const int b = t;
    float mx = -1e30f;
    for (int p = 0; p < 8; ++p) mx = fmaxf(mx, sc[p * 8 + b]);
    float e[8]; float se = 0.f;
    for (int p = 0; p < 8; ++p) { e[p] = expf(sc[p * 8 + b] - mx); se += e[p]; }
    float a[8]; float s2 = 0.f;
    for (int p = 0; p < 8; ++p) { a[p] = e[p] / se; if (a[p] < 0.1f) a[p] = 0.f; s2 += a[p]; }
    const float inv = 1.0f / (s2 + 1e-6f);
    for (int p = 0; p < 8; ++p) amps[p * 8 + b] = a[p] * inv;
  }
}

// ---------------- collapse paths ----------------
__global__ __launch_bounds__(256) void k_final(const float* __restrict__ amps,
                                               const float* __restrict__ th,
                                               float* __restrict__ fin) {
  const int idx = blockIdx.x * 256 + threadIdx.x;
  const int b = idx >> 10;
  const int d = idx & (D_DIM - 1);
  float s = 0.f;
#pragma unroll
  for (int p = 0; p < 8; ++p)
    s += amps[p * 8 + b] * th[(size_t)(p * 8 + b) * D_DIM + d];
  fin[idx] = s;
}

// ---------------- fused gate GEMM + sigmoid gate + residual + LayerNorm ----------------
// One block per 64 rows of [B*L, D]. Full N=1024 via 4 register-tiled n-chunks.
// B operand (bf16 Wt chunk, 256 n x 32 k) is DMA'd into LDS by the Tensor Data
// Mover (double-buffered, s_wait_tensorcnt pipelined); TDM pads 16 B after each
// 64 B row so the LDS stride (40 bf16) breaks bank conflicts. A operand (h) is
// converted fp32->bf16 into LDS by the waves. LN stats are accumulated in
// registers (unique slots, no atomics -> deterministic), unnormalized rows are
// written to out and normalized in place.
#define KC 32
#define ASTRIDE 40
#define BSTRIDE 40
#define BS_ELEMS (256 * BSTRIDE)

__global__ __launch_bounds__(256) void k_gate(const float* __restrict__ h,
                                              const unsigned short* __restrict__ Wtg,
                                              const float* __restrict__ bg,
                                              const float* __restrict__ bc,
                                              const float* __restrict__ gamma,
                                              const float* __restrict__ beta,
                                              float* __restrict__ out) {
  __shared__ __align__(16) unsigned short As[64 * ASTRIDE];     //  5.0 KB
  __shared__ __align__(16) unsigned short Bs[2 * BS_ELEMS];     // 40.0 KB
  __shared__ float rsum[64 * 32];                               //  8 KB
  __shared__ float rsq [64 * 32];                               //  8 KB
  __shared__ float mu_s[64];
  __shared__ float rstd_s[64];

  const int t = threadIdx.x;
  const int lane = t & 31;
  const int w = t >> 5;
  const int m = (w & 3) * 16;       // local m tile
  const int nhid = w >> 2;          // 0/1
  const int nh = nhid * 128;        // n offset within 256-wide super chunk
  const int row0 = blockIdx.x * 64; // global row base
  const int b = row0 >> 11;         // row / 2048
  const float* hrow = h + (size_t)row0 * D_DIM;
  const int rbase = m + ((lane & 16) ? 8 : 0);
  const int cl = lane & 15;

  float rs_loc[8], rq_loc[8];
#pragma unroll
  for (int v = 0; v < 8; ++v) { rs_loc[v] = 0.f; rq_loc[v] = 0.f; }

  for (int ns = 0; ns < 4; ++ns) {
    const int nsup = ns * 256;
    f32x8 acc[8];
#pragma unroll
    for (int j = 0; j < 8; ++j) acc[j] = zero8();

    __syncthreads();   // previous n-super's compute fully done before reuse of Bs
#if HAVE_TDM
    if (t == 0)
      tdm_issue_b(Wtg + (size_t)nsup * D_DIM, (unsigned)(size_t)(void*)&Bs[0]);
#endif
    int buf = 0;
    for (int kc = 0; kc < D_DIM / KC; ++kc) {   // 32 chunks
      const int kb = kc * KC;
      __syncthreads();   // all waves done with compute of chunk kc-1
#if HAVE_TDM
      if (t == 0) {
        if (kc + 1 < D_DIM / KC) {
          tdm_issue_b(Wtg + (size_t)nsup * D_DIM + (kb + KC),
                      (unsigned)(size_t)(void*)&Bs[(buf ^ 1) * BS_ELEMS]);
          __builtin_amdgcn_s_wait_tensorcnt(1);   // chunk kc resident (in-order)
        } else {
          __builtin_amdgcn_s_wait_tensorcnt(0);
        }
      }
#else
      // synchronous fallback staging of B chunk into Bs[buf]
      for (int i = t; i < 256 * 4; i += 256) {
        const int n = i >> 2;
        const int c8 = (i & 3) * 8;
        const unsigned short* src = Wtg + (size_t)(nsup + n) * D_DIM + kb + c8;
        *(uint4*)(Bs + buf * BS_ELEMS + n * BSTRIDE + c8) = *(const uint4*)src;
        if (kc + 1 < D_DIM / KC) __builtin_prefetch(src + KC, 0, 1);
      }
#endif
      // stage A chunk: h fp32 -> bf16 LDS (64 rows x 32 k)
      for (int i = t; i < 64 * 8; i += 256) {
        const int r = i >> 3;
        const int c4 = (i & 7) * 4;
        float4 f = *(const float4*)(hrow + (size_t)r * D_DIM + kb + c4);
        const int o = r * ASTRIDE + c4;
        As[o + 0] = f2bf(f.x); As[o + 1] = f2bf(f.y);
        As[o + 2] = f2bf(f.z); As[o + 3] = f2bf(f.w);
      }
      __syncthreads();   // As staged; Bs[buf] TDM-complete published to all waves
      bf16x16 a = load_a_frag(As, ASTRIDE, m, 0, lane);
#pragma unroll
      for (int j = 0; j < 8; ++j) {
        bf16x16 bb = load_b_frag(Bs + buf * BS_ELEMS, BSTRIDE, nh + j * 16, 0, lane);
        acc[j] = wmma_bf16f32(a, bb, acc[j]);
      }
      buf ^= 1;
    }
    // epilogue: bias + sigmoid gate + residual; accumulate LN stats in regs
    for (int j = 0; j < 8; ++j) {
      const int col = nsup + nh + j * 16 + cl;
      const float bgv = bg[col];
      const float bcv = bc[(size_t)b * D_DIM + col];
#pragma unroll
      for (int v = 0; v < 8; ++v) {
        const int rl = rbase + v;
        const float x = acc[j][v] + bgv;
        const float g = 1.0f / (1.0f + expf(-x));
        const float hv = hrow[(size_t)rl * D_DIM + col];
        const float o = hv + g * bcv;
        out[(size_t)(row0 + rl) * D_DIM + col] = o;
        rs_loc[v] += o;
        rq_loc[v] += o * o;
      }
    }
  }
  // unique-slot stat stores (no atomics -> deterministic)
#pragma unroll
  for (int v = 0; v < 8; ++v) {
    rsum[(rbase + v) * 32 + cl * 2 + nhid] = rs_loc[v];
    rsq [(rbase + v) * 32 + cl * 2 + nhid] = rq_loc[v];
  }
  asm volatile("s_wait_storecnt 0x0" ::: "memory");  // block's global stores visible
  __syncthreads();
  if (t < 64) {
    float s = 0.f, q = 0.f;
    for (int i = 0; i < 32; ++i) { s += rsum[t * 32 + i]; q += rsq[t * 32 + i]; }
    const float mu = s * (1.0f / D_DIM);
    const float var = q * (1.0f / D_DIM) - mu * mu;
    mu_s[t] = mu;
    rstd_s[t] = rsqrtf(var + 1e-6f);
  }
  __syncthreads();
  // in-place normalize the block's 64 rows
  for (int i = t; i < 64 * 256; i += 256) {
    const int r = i >> 8;
    const int c4 = (i & 255) * 4;
    const size_t off = (size_t)(row0 + r) * D_DIM + c4;
    float4 x = *(const float4*)(out + off);
    const float4 gm = *(const float4*)(gamma + c4);
    const float4 bt = *(const float4*)(beta + c4);
    const float mu = mu_s[r], rstd = rstd_s[r];
    x.x = (x.x - mu) * rstd * gm.x + bt.x;
    x.y = (x.y - mu) * rstd * gm.y + bt.y;
    x.z = (x.z - mu) * rstd * gm.z + bt.z;
    x.w = (x.w - mu) * rstd * gm.w + bt.w;
    *(float4*)(out + off) = x;
  }
}

// ---------------------------------------------------------------------------
// host launch
// ---------------------------------------------------------------------------
extern "C" void kernel_launch(void* const* d_in, const int* in_sizes, int n_in,
                              void* d_out, int out_size, void* d_ws, size_t ws_size,
                              hipStream_t stream) {
  const float* h    = (const float*)d_in[0];
  const float* ing  = (const float*)d_in[1];
  const float* inb  = (const float*)d_in[2];
  const float* Wa   = (const float*)d_in[3];
  const float* ba   = (const float*)d_in[4];
  const float* png  = (const float*)d_in[5];
  const float* pnb  = (const float*)d_in[6];
  const float* W1   = (const float*)d_in[7];
  const float* b1   = (const float*)d_in[8];
  const float* W2   = (const float*)d_in[9];
  const float* b2   = (const float*)d_in[10];
  const float* Wbc  = (const float*)d_in[11];
  const float* bbc  = (const float*)d_in[12];
  const float* Wg   = (const float*)d_in[13];
  const float* bg   = (const float*)d_in[14];
  const float* ong  = (const float*)d_in[15];
  const float* onb  = (const float*)d_in[16];
  float* out = (float*)d_out;
  char* ws = (char*)d_ws;

  // workspace layout (bytes)
  size_t off = 0;
  unsigned short* Wtg = (unsigned short*)(ws + off); off += (size_t)D_DIM * D_DIM * 2;  // 2 MB
  unsigned short* Wt1 = (unsigned short*)(ws + off); off += (size_t)H_DIM * D_DIM * 2;  // 8 MB
  unsigned short* Wt2 = (unsigned short*)(ws + off); off += (size_t)D_DIM * H_DIM * 2;  // 8 MB
  float* partial  = (float*)(ws + off); off += (size_t)B_DIM * 32 * D_DIM * 4;          // 1 MB
  float* context  = (float*)(ws + off); off += (size_t)B_DIM * D_DIM * 4;
  float* ctxn     = (float*)(ws + off); off += (size_t)B_DIM * D_DIM * 4;
  float* thoughts = (float*)(ws + off); off += (size_t)NROWS * D_DIM * 4;
  unsigned short* xln = (unsigned short*)(ws + off); off += (size_t)NROWS * D_DIM * 2;
  unsigned short* y1b = (unsigned short*)(ws + off); off += (size_t)NROWS * H_DIM * 2;
  float* amps     = (float*)(ws + off); off += 256;
  float* finalv   = (float*)(ws + off); off += (size_t)B_DIM * D_DIM * 4;
  float* bcv      = (float*)(ws + off); off += (size_t)B_DIM * D_DIM * 4;
  (void)off; (void)in_sizes; (void)n_in; (void)out_size; (void)ws_size;

  // 1) one-time weight transpose/convert to bf16 [N][K]
  k_convw<<<dim3(D_DIM / 32, D_DIM / 32), 256, 0, stream>>>(Wg, Wtg, D_DIM, D_DIM);
  k_convw<<<dim3(H_DIM / 32, D_DIM / 32), 256, 0, stream>>>(W1, Wt1, D_DIM, H_DIM);
  k_convw<<<dim3(D_DIM / 32, H_DIM / 32), 256, 0, stream>>>(W2, Wt2, H_DIM, D_DIM);

  // 2) context = mean over L
  k_ctx_partial<<<dim3(32, B_DIM), 256, 0, stream>>>(h, partial);
  k_ctx_reduce<<<B_DIM, 256, 0, stream>>>(partial, context);

  // 3) LN(context) -> aggregate -> seed 8 paths
  k_ln_rows<false><<<B_DIM, 256, 0, stream>>>(context, ing, inb, (void*)ctxn);
  k_agg_seed<<<32, 256, 0, stream>>>(ctxn, Wa, ba, thoughts);

  // 4) four thought steps
  for (int s = 0; s < 4; ++s) {
    k_ln_rows<true><<<NROWS, 256, 0, stream>>>(thoughts, png, pnb, (void*)xln);
    k_fc1<<<H_DIM / 32, 256, 0, stream>>>(xln, Wt1, b1, y1b);
    k_fc2<<<D_DIM / 32, 256, 0, stream>>>(y1b, Wt2, b2, thoughts);
    k_score<<<1, 256, 0, stream>>>(thoughts, context, amps);
  }

  // 5) collapse paths, broadcast projection
  k_final<<<32, 256, 0, stream>>>(amps, thoughts, finalv);
  k_bcast<<<32, 256, 0, stream>>>(finalv, Wbc, bbc, bcv);

  // 6) fused gate GEMM + sigmoid + residual + LayerNorm (TDM double-buffered)
  k_gate<<<(B_DIM * L_DIM) / 64, 256, 0, stream>>>(h, Wtg, bg, bcv, ong, onb, out);
}
